// DAGERC_multimodal_67731634258669
// MI455X (gfx1250) — matrix operand order repaired and checked by reading, not compiled, not run on hardware
//
#include <hip/hip_runtime.h>

typedef __bf16 bf16_t;
typedef __attribute__((ext_vector_type(16))) __bf16 v16bf;
typedef __attribute__((ext_vector_type(8)))  float   v8f;

#define Bb     16
#define Nn     80
#define BN     1280
#define TD     1024
#define AD     512
#define FD     512
#define FO     512
#define DH     384
#define DH3    1152
#define NCc    7
#define INDIM  1664

__device__ __forceinline__ float sigf(float x) { return 1.0f / (1.0f + __expf(-x)); }

// ---- WMMA fragment loaders (bf16, 16x16x32; CDNA5 wave32 layouts) ----
// A 16x32 (MxK): lane<16 -> row M=lane, K = kk+{0..7, 16..23}
//                lane>=16 -> row M=lane-16, K = kk+{8..15, 24..31}
__device__ __forceinline__ v16bf load_a_frag(const bf16_t* A, int lda, int m0, int kk, int lane) {
  int half = (lane >> 4) & 1;
  int lr   = lane & 15;
  const bf16_t* p = A + (size_t)(m0 + lr) * lda + kk + half * 8;
  v16bf a;
#pragma unroll
  for (int t = 0; t < 8; ++t) { a[t] = p[t]; a[t + 8] = p[t + 16]; }
  return a;
}
// B 32x16 (KxN), sourced from weight W[n][k] (row-major, ld = K):
// lane<16 -> col N=lane, K = kk+{0..7, 16..23} ; lane>=16 -> col N=lane-16, K = kk+{8..15, 24..31}
__device__ __forceinline__ v16bf load_b_frag(const bf16_t* W, int ldw, int n0, int kk, int lane) {
  int half = (lane >> 4) & 1;
  int lr   = lane & 15;
  const bf16_t* p = W + (size_t)(n0 + lr) * ldw + kk + half * 8;
  v16bf b;
#pragma unroll
  for (int t = 0; t < 8; ++t) { b[t] = p[t]; b[t + 8] = p[t + 16]; }
  return b;
}

// ---- wave tile GEMM core. DUAL is compile-time: no EXEC-masked WMMA possible.
// K must be a multiple of 64 (true for all K used: 384/512/1024/1664).
// Two k-chunks of loads are issued before consumption -> load/WMMA overlap;
// with DUAL, two independent accumulator chains share each A fragment.
template<bool DUAL>
__device__ __forceinline__ void gemm_tile(const bf16_t* __restrict__ A, int lda, int m0,
                                          const bf16_t* __restrict__ W, int ldw, int n0,
                                          int K, int lane, v8f& acc0, v8f& acc1) {
  int lr = lane & 15;
  for (int kk = 0; kk < K; kk += 64) {
    v16bf a0  = load_a_frag(A, lda, m0, kk, lane);
    v16bf b00 = load_b_frag(W, ldw, n0, kk, lane);
    v16bf b10 = {};
    if (DUAL) b10 = load_b_frag(W, ldw, n0 + 16, kk, lane);
    v16bf a1  = load_a_frag(A, lda, m0, kk + 32, lane);
    v16bf b01 = load_b_frag(W, ldw, n0, kk + 32, lane);
    v16bf b11 = {};
    if (DUAL) b11 = load_b_frag(W, ldw, n0 + 16, kk + 32, lane);
    if (kk + 64 < K) {
      __builtin_prefetch(A + (size_t)(m0 + lr) * lda + kk + 64, 0, 1);
      __builtin_prefetch(W + (size_t)(n0 + lr) * ldw + kk + 64, 0, 1);
    }
    acc0 = __builtin_amdgcn_wmma_f32_16x16x32_bf16(false, a0, false, b00, (short)0, acc0, false, false);
    if (DUAL)
      acc1 = __builtin_amdgcn_wmma_f32_16x16x32_bf16(false, a0, false, b10, (short)0, acc1, false, false);
    acc0 = __builtin_amdgcn_wmma_f32_16x16x32_bf16(false, a1, false, b01, (short)0, acc0, false, false);
    if (DUAL)
      acc1 = __builtin_amdgcn_wmma_f32_16x16x32_bf16(false, a1, false, b11, (short)0, acc1, false, false);
  }
}

// ---- fp32 -> bf16 conversion ----
__global__ void k_cvt(const float* __restrict__ s, bf16_t* __restrict__ d, int n) {
  int i = blockIdx.x * blockDim.x + threadIdx.x;
  if (i < n) d[i] = (bf16_t)s[i];
}
__global__ void k_cvt_pad(const float* __restrict__ s, bf16_t* __restrict__ d,
                          int rows, int cols, int padrows) {
  int i = blockIdx.x * blockDim.x + threadIdx.x;
  int n = padrows * cols;
  if (i >= n) return;
  int r = i / cols;
  d[i] = (r < rows) ? (bf16_t)s[i] : (bf16_t)0.0f;
}

// ---- generic WMMA GEMM: C = act( A @ W^T + bias [+ addin] ) ----
// act: 0=none, 1=relu, 2=sigmoid, 3 = sigmoid(x) * addin
// DUAL=true: each wave computes a 16x32 tile; DUAL=false: 16x16 tile.
template<bool DUAL>
__global__ void k_gemm(const bf16_t* __restrict__ A, int lda,
                       const bf16_t* __restrict__ Wt, int ldw,
                       const float* __restrict__ bias,
                       const float* __restrict__ addin,
                       float* __restrict__ C, int ldc,
                       int M, int Npad, int Nreal, int K, int act) {
  int lane = threadIdx.x & 31;
  int wid  = threadIdx.x >> 5;
  int tile = blockIdx.x * (blockDim.x >> 5) + wid;
  const int nw = DUAL ? 32 : 16;
  int ntP  = Npad / nw;
  int ntiles = (M >> 4) * ntP;
  if (tile >= ntiles) return;                // wave-uniform exit (EXEC all-ones for WMMA)
  int m0 = (tile / ntP) << 4;
  int n0 = (tile % ntP) * nw;
  v8f acc0 = {}, acc1 = {};
  gemm_tile<DUAL>(A, lda, m0, Wt, ldw, n0, K, lane, acc0, acc1);

  int half = (lane >> 4) & 1;
  int lr   = lane & 15;
#pragma unroll
  for (int s = 0; s < (DUAL ? 2 : 1); ++s) {
    int col = n0 + s * 16 + lr;
    if (col >= Nreal) continue;              // store-side predication only
    const v8f& acc = s ? acc1 : acc0;
    float bv = bias ? bias[col] : 0.0f;
#pragma unroll
    for (int r = 0; r < 8; ++r) {
      int row = m0 + r + half * 8;           // C/D layout: VGPR r -> M=r (lanes<16), M=r+8 (lanes>=16)
      float v = acc[r] + bv;
      if (act == 3)      v = sigf(v) * addin[(size_t)row * ldc + col];
      else if (addin)    v += addin[(size_t)row * ldc + col];
      if (act == 1)      v = fmaxf(v, 0.0f);
      else if (act == 2) v = sigf(v);
      C[(size_t)row * ldc + col] = v;
    }
  }
}

// ---- out[r] = A[r,:] . v ----
__global__ void k_rowdot(const float* __restrict__ A, const float* __restrict__ v,
                         float* __restrict__ out, int rows, int k) {
  int r = blockIdx.x * blockDim.x + threadIdx.x;
  if (r >= rows) return;
  float s = 0.0f;
  for (int d = 0; d < k; ++d) s += A[(size_t)r * k + d] * v[d];
  out[r] = s;
}

// ---- sequential DAG-GRU scan for one layer; single workgroup, 16 waves ----
__global__ __launch_bounds__(512)
void k_scan(const float* __restrict__ Hin,   // (BN, DH) layer input (xi rows)
            float* __restrict__ Hout,        // (BN, DH) layer output
            const float* __restrict__ Q,     // (BN)  Hin . wq
            const float* __restrict__ GIC,   // (BN, 3DH)  Hin@cwi^T + cbi
            const float* __restrict__ GHP,   // (BN, 3DH)  Hin@pwh^T + pbh
            float* __restrict__ Apr,         // (BN, DH)  Hnew@wr0^T
            float* __restrict__ Bpr,         // (BN, DH)  Hnew@wr1^T
            float* __restrict__ ghC,         // (16, 3DH) scratch
            float* __restrict__ giP,         // (16, 3DH) scratch
            float* __restrict__ Mg,          // (16, DH)  scratch
            const bf16_t* __restrict__ cwhb, const bf16_t* __restrict__ pwib,
            const bf16_t* __restrict__ wr0b, const bf16_t* __restrict__ wr1b,
            const float* __restrict__ cbh,   const float* __restrict__ pbi,
            const float* __restrict__ gatw,  const float* __restrict__ gatb, int layer,
            const int* __restrict__ adj,     const int* __restrict__ smask) {
  __shared__ float  hk[Bb * Nn];
  __shared__ float  alf[Bb * Nn];
  __shared__ float  wkv[DH];
  __shared__ bf16_t Mbf[Bb * DH];
  __shared__ bf16_t Hbf[Bb * DH];

  const int tid  = threadIdx.x;
  const int lane = tid & 31;
  const int wid  = tid >> 5;          // 0..15
  const float gb = gatb[layer];
  const float* wk = gatw + layer * 2 * DH + DH;

  for (int t = tid; t < DH; t += 512) wkv[t] = wk[t];
  __syncthreads();

  for (int i = 0; i < Nn; ++i) {
    // ---- phase A: attention weights + context M ----
    if (i == 0) {
      for (int t = tid; t < Bb * DH; t += 512) { Mg[t] = 0.0f; Mbf[t] = (bf16_t)0.0f; }
    } else {
      int cnt = Bb * i;
      for (int t = tid; t < cnt; t += 512) {
        int b = t / i, j = t - b * i;
        float a = Q[b * Nn + i] + hk[b * Nn + j] + gb;
        if (adj[b * Nn * Nn + i * Nn + j] == 0) a = -1e30f;
        alf[b * Nn + j] = a;
      }
      __syncthreads();
      if (tid < Bb) {
        int b = tid;
        float mx = -1e30f;
        for (int j = 0; j < i; ++j) mx = fmaxf(mx, alf[b * Nn + j]);
        float s = 0.0f;
        for (int j = 0; j < i; ++j) {
          float e = (mx > -1e29f) ? __expf(alf[b * Nn + j] - mx) : 0.0f;
          alf[b * Nn + j] = e; s += e;
        }
        float inv = (s > 0.0f) ? 1.0f / s : 0.0f;
        for (int j = 0; j < i; ++j) alf[b * Nn + j] *= inv;
      }
      __syncthreads();
      for (int t = tid; t < Bb * DH; t += 512) {
        int b = t / DH, d = t - b * DH;
        float acc = 0.0f;
        for (int j = 0; j < i; ++j) {
          float w = alf[b * Nn + j];
          if (w != 0.0f) {
            int sm = smask[b * Nn * Nn + i * Nn + j];
            float v = sm ? Apr[((size_t)b * Nn + j) * DH + d]
                         : Bpr[((size_t)b * Nn + j) * DH + d];
            acc += w * v;
          }
        }
        Mg[t] = acc;
        Mbf[t] = (bf16_t)acc;
      }
    }
    __threadfence(); __syncthreads();

    // ---- phase B: gh_C = M@cwh^T + cbh ; gi_P = M@pwi^T + pbi (WMMA, 72 pair-tiles) ----
    for (int p = wid; p < 2 * (DH3 / 32); p += 16) {
      bool isC = p < (DH3 / 32);
      int  n0  = (isC ? p : p - DH3 / 32) << 5;
      const bf16_t* Wb = isC ? cwhb : pwib;
      const float*  bs = isC ? cbh : pbi;
      float*        Og = isC ? ghC : giP;
      v8f acc0 = {}, acc1 = {};
      gemm_tile<true>(Mbf, DH, 0, Wb, DH, n0, DH, lane, acc0, acc1);
      int half = (lane >> 4) & 1, lr = lane & 15;
#pragma unroll
      for (int s = 0; s < 2; ++s) {
        int col = n0 + s * 16 + lr;
        const v8f& acc = s ? acc1 : acc0;
        float bv = bs[col];
#pragma unroll
        for (int r = 0; r < 8; ++r) Og[(r + half * 8) * DH3 + col] = acc[r] + bv;
      }
    }
    __threadfence(); __syncthreads();

    // ---- phase C: GRU gates, Hnew = C + P ----
    for (int t = tid; t < Bb * DH; t += 512) {
      int b = t / DH, d = t - b * DH;
      size_t rg = ((size_t)b * Nn + i) * DH3;
      float icr = GIC[rg + d], icz = GIC[rg + DH + d], icn = GIC[rg + 2 * DH + d];
      float hcr = ghC[b * DH3 + d], hcz = ghC[b * DH3 + DH + d], hcn = ghC[b * DH3 + 2 * DH + d];
      float r1 = sigf(icr + hcr), z1 = sigf(icz + hcz);
      float n1 = tanhf(icn + r1 * hcn);
      float mv = Mg[t];
      float Cv = (1.0f - z1) * n1 + z1 * mv;
      float pir = giP[b * DH3 + d], piz = giP[b * DH3 + DH + d], pin = giP[b * DH3 + 2 * DH + d];
      float phr = GHP[rg + d], phz = GHP[rg + DH + d], phn = GHP[rg + 2 * DH + d];
      float r2 = sigf(pir + phr), z2 = sigf(piz + phz);
      float n2 = tanhf(pin + r2 * phn);
      float xv = Hin[((size_t)b * Nn + i) * DH + d];
      float Pv = (1.0f - z2) * n2 + z2 * xv;
      float hn = Cv + Pv;
      Hout[((size_t)b * Nn + i) * DH + d] = hn;
      Hbf[t] = (bf16_t)hn;
    }
    __syncthreads();

    // ---- phase D: Apr/Bpr = Hnew @ wr{0,1}^T (WMMA, 24 pair-tiles) + hk update ----
    for (int p = wid; p < 2 * (DH / 32); p += 16) {
      bool isA = p < (DH / 32);
      int  n0  = (isA ? p : p - DH / 32) << 5;
      const bf16_t* Wb = isA ? wr0b : wr1b;
      float*        Og = isA ? Apr : Bpr;
      v8f acc0 = {}, acc1 = {};
      gemm_tile<true>(Hbf, DH, 0, Wb, DH, n0, DH, lane, acc0, acc1);
      int half = (lane >> 4) & 1, lr = lane & 15;
#pragma unroll
      for (int s = 0; s < 2; ++s) {
        int col = n0 + s * 16 + lr;
        const v8f& acc = s ? acc1 : acc0;
#pragma unroll
        for (int r = 0; r < 8; ++r) {
          int b = r + half * 8;
          Og[((size_t)b * Nn + i) * DH + col] = acc[r];
        }
      }
    }
    if (tid < Bb) {
      int b = tid; float s = 0.0f;
      for (int d = 0; d < DH; ++d) s += (float)Hbf[b * DH + d] * wkv[d];
      hk[b * Nn + i] = s;
    }
    __threadfence(); __syncthreads();
  }
}

// ---- Hcat = [H0 | H1 | H2 | fused] as bf16 ----
__global__ void k_concat(const float* __restrict__ H0, const float* __restrict__ H1,
                         const float* __restrict__ H2, const float* __restrict__ fu,
                         bf16_t* __restrict__ out) {
  int idx = blockIdx.x * blockDim.x + threadIdx.x;
  if (idx >= BN * INDIM) return;
  int r = idx / INDIM, c = idx - r * INDIM;
  float v;
  if (c < DH)              v = H0[(size_t)r * DH + c];
  else if (c < 2 * DH)     v = H1[(size_t)r * DH + c - DH];
  else if (c < 3 * DH)     v = H2[(size_t)r * DH + c - 2 * DH];
  else                     v = fu[(size_t)r * FO + c - 3 * DH];
  out[idx] = (bf16_t)v;
}

__global__ void k_logsoftmax(const float* __restrict__ lg, float* __restrict__ out, int rows) {
  int r = blockIdx.x * blockDim.x + threadIdx.x;
  if (r >= rows) return;
  float mx = -1e30f;
  for (int c = 0; c < NCc; ++c) mx = fmaxf(mx, lg[r * NCc + c]);
  float s = 0.0f;
  for (int c = 0; c < NCc; ++c) s += __expf(lg[r * NCc + c] - mx);
  float ls = logf(s);
  for (int c = 0; c < NCc; ++c) out[r * NCc + c] = lg[r * NCc + c] - mx - ls;
}

extern "C" void kernel_launch(void* const* d_in, const int* in_sizes, int n_in,
                              void* d_out, int out_size, void* d_ws, size_t ws_size,
                              hipStream_t stream) {
  (void)in_sizes; (void)n_in; (void)out_size; (void)ws_size;
  const float* f_text = (const float*)d_in[0];
  const float* f_aud  = (const float*)d_in[1];
  const int*   adj    = (const int*)d_in[2];
  const int*   smask  = (const int*)d_in[3];
  const float* U1 = (const float*)d_in[6];
  const float* U2 = (const float*)d_in[7];
  const float* Pw = (const float*)d_in[8];
  const float* Pbias = (const float*)d_in[9];
  const float* V1 = (const float*)d_in[10];
  const float* V2 = (const float*)d_in[11];
  const float* fc1w = (const float*)d_in[12];
  const float* fc1b = (const float*)d_in[13];
  const float* gatw = (const float*)d_in[14];
  const float* gatb = (const float*)d_in[15];
  const float* Wr0 = (const float*)d_in[16];
  const float* Wr1 = (const float*)d_in[17];
  const float* gCwi = (const float*)d_in[18];
  const float* gCwh = (const float*)d_in[19];
  const float* gCbi = (const float*)d_in[20];
  const float* gCbh = (const float*)d_in[21];
  const float* gPwi = (const float*)d_in[22];
  const float* gPwh = (const float*)d_in[23];
  const float* gPbi = (const float*)d_in[24];
  const float* gPbh = (const float*)d_in[25];
  const float* w0 = (const float*)d_in[26]; const float* b0 = (const float*)d_in[27];
  const float* w1 = (const float*)d_in[28]; const float* b1 = (const float*)d_in[29];
  const float* w2 = (const float*)d_in[30]; const float* b2 = (const float*)d_in[31];

  char* base = (char*)d_ws;
  size_t cur = 0;
  auto alloc = [&](size_t bytes) -> void* {
    cur = (cur + 255) & ~(size_t)255;
    void* p = base + cur; cur += bytes; return p;
  };
  // fp32 scratch
  float* cbuf = (float*)alloc((size_t)BN * FD * 4);
  float* fused = (float*)alloc((size_t)BN * FO * 4);
  float* H0  = (float*)alloc((size_t)BN * DH * 4);
  float* H1o = (float*)alloc((size_t)BN * DH * 4);
  float* H2o = (float*)alloc((size_t)BN * DH * 4);
  float* Qb  = (float*)alloc((size_t)BN * 4);
  float* GIC = (float*)alloc((size_t)BN * DH3 * 4);
  float* GHP = (float*)alloc((size_t)BN * DH3 * 4);
  float* Apr = (float*)alloc((size_t)BN * DH * 4);
  float* Bpr = (float*)alloc((size_t)BN * DH * 4);
  float* ghC = (float*)alloc((size_t)Bb * DH3 * 4);
  float* giP = (float*)alloc((size_t)Bb * DH3 * 4);
  float* Mg  = (float*)alloc((size_t)Bb * DH * 4);
  float* h1  = (float*)alloc((size_t)BN * DH * 4);
  float* h2  = (float*)alloc((size_t)BN * DH * 4);
  float* lg  = (float*)alloc((size_t)BN * NCc * 4);
  // bf16 activations
  bf16_t* tbf   = (bf16_t*)alloc((size_t)BN * TD * 2);
  bf16_t* abf   = (bf16_t*)alloc((size_t)BN * AD * 2);
  bf16_t* cbbf  = (bf16_t*)alloc((size_t)BN * FD * 2);
  bf16_t* fubf  = (bf16_t*)alloc((size_t)BN * FO * 2);
  bf16_t* Hinbf = (bf16_t*)alloc((size_t)BN * DH * 2);
  bf16_t* Hcatb = (bf16_t*)alloc((size_t)BN * INDIM * 2);
  bf16_t* h1bf  = (bf16_t*)alloc((size_t)BN * DH * 2);
  bf16_t* h2bf  = (bf16_t*)alloc((size_t)BN * DH * 2);
  // bf16 weights
  bf16_t* U1b = (bf16_t*)alloc((size_t)FD * TD * 2);
  bf16_t* U2b = (bf16_t*)alloc((size_t)FD * AD * 2);
  bf16_t* Pb2 = (bf16_t*)alloc((size_t)FO * FD * 2);
  bf16_t* V1b = (bf16_t*)alloc((size_t)FO * TD * 2);
  bf16_t* V2b = (bf16_t*)alloc((size_t)FO * AD * 2);
  bf16_t* fc1bb = (bf16_t*)alloc((size_t)DH * FO * 2);
  bf16_t* cwib = (bf16_t*)alloc((size_t)DH3 * DH * 2);
  bf16_t* cwhb = (bf16_t*)alloc((size_t)DH3 * DH * 2);
  bf16_t* pwib = (bf16_t*)alloc((size_t)DH3 * DH * 2);
  bf16_t* pwhb = (bf16_t*)alloc((size_t)DH3 * DH * 2);
  bf16_t* wr0b = (bf16_t*)alloc((size_t)DH * DH * 2);
  bf16_t* wr1b = (bf16_t*)alloc((size_t)DH * DH * 2);
  bf16_t* w0b = (bf16_t*)alloc((size_t)DH * INDIM * 2);
  bf16_t* w1b = (bf16_t*)alloc((size_t)DH * DH * 2);
  bf16_t* w2b = (bf16_t*)alloc((size_t)16 * DH * 2);

  auto cvt = [&](const float* s, bf16_t* d, size_t n) {
    k_cvt<<<(int)((n + 255) / 256), 256, 0, stream>>>(s, d, (int)n);
  };
  auto gemm = [&](const bf16_t* A, int lda, const bf16_t* Wt, int ldw, const float* bias,
                  const float* addin, float* C, int ldc, int M, int Npad, int Nreal, int K, int act) {
    int tiles = (M / 16) * (Npad / 32);
    k_gemm<true><<<(tiles + 7) / 8, 256, 0, stream>>>(A, lda, Wt, ldw, bias, addin, C, ldc,
                                                      M, Npad, Nreal, K, act);
  };
  auto gemm1 = [&](const bf16_t* A, int lda, const bf16_t* Wt, int ldw, const float* bias,
                   float* C, int ldc, int M, int Npad, int Nreal, int K) {
    int tiles = (M / 16) * (Npad / 16);
    k_gemm<false><<<(tiles + 7) / 8, 256, 0, stream>>>(A, lda, Wt, ldw, bias, nullptr, C, ldc,
                                                       M, Npad, Nreal, K, 0);
  };

  // weights -> bf16 (once)
  cvt(U1, U1b, (size_t)FD * TD);   cvt(U2, U2b, (size_t)FD * AD);
  cvt(Pw, Pb2, (size_t)FO * FD);   cvt(V1, V1b, (size_t)FO * TD);
  cvt(V2, V2b, (size_t)FO * AD);   cvt(fc1w, fc1bb, (size_t)DH * FO);
  cvt(w0, w0b, (size_t)DH * INDIM); cvt(w1, w1b, (size_t)DH * DH);
  k_cvt_pad<<<(16 * DH + 255) / 256, 256, 0, stream>>>(w2, w2b, NCc, DH, 16);
  // activations -> bf16
  cvt(f_text, tbf, (size_t)BN * TD);
  cvt(f_aud,  abf, (size_t)BN * AD);

  // frontend: c_ = sig(text@U1^T) * sig(audio@U2^T)
  gemm(tbf, TD, U1b, TD, nullptr, nullptr, cbuf, FD, BN, FD, FD, TD, 2);
  gemm(abf, AD, U2b, AD, nullptr, cbuf,    cbuf, FD, BN, FD, FD, AD, 3);
  cvt(cbuf, cbbf, (size_t)BN * FD);
  // fused = c@P^T + P_b + text@V1^T + audio@V2^T
  gemm(cbbf, FD, Pb2, FD, Pbias, nullptr, fused, FO, BN, FO, FO, FD, 0);
  gemm(tbf,  TD, V1b, TD, nullptr, fused, fused, FO, BN, FO, FO, TD, 0);
  gemm(abf,  AD, V2b, AD, nullptr, fused, fused, FO, BN, FO, FO, AD, 0);
  cvt(fused, fubf, (size_t)BN * FO);
  // H0 = relu(fused@fc1^T + fc1_b)
  gemm(fubf, FO, fc1bb, FO, fc1b, nullptr, H0, DH, BN, DH, DH, FO, 1);

  // two sequential DAG-GRU layers
  for (int l = 0; l < 2; ++l) {
    const float* Hin = (l == 0) ? H0 : H1o;
    float*       Ho  = (l == 0) ? H1o : H2o;
    cvt(gCwi + (size_t)l * DH3 * DH, cwib, (size_t)DH3 * DH);
    cvt(gCwh + (size_t)l * DH3 * DH, cwhb, (size_t)DH3 * DH);
    cvt(gPwi + (size_t)l * DH3 * DH, pwib, (size_t)DH3 * DH);
    cvt(gPwh + (size_t)l * DH3 * DH, pwhb, (size_t)DH3 * DH);
    cvt(Wr0 + (size_t)l * DH * DH, wr0b, (size_t)DH * DH);
    cvt(Wr1 + (size_t)l * DH * DH, wr1b, (size_t)DH * DH);
    cvt(Hin, Hinbf, (size_t)BN * DH);
    // hoisted batch work: Q = Hin.wq ; GI_C = Hin@cwi^T+cbi ; GH_P = Hin@pwh^T+pbh
    k_rowdot<<<(BN + 255) / 256, 256, 0, stream>>>(Hin, gatw + l * 2 * DH, Qb, BN, DH);
    gemm(Hinbf, DH, cwib, DH, gCbi + l * DH3, nullptr, GIC, DH3, BN, DH3, DH3, DH, 0);
    gemm(Hinbf, DH, pwhb, DH, gPbh + l * DH3, nullptr, GHP, DH3, BN, DH3, DH3, DH, 0);
    // sequential scan (persistent workgroup, 16 waves, WMMA inside)
    k_scan<<<1, 512, 0, stream>>>(Hin, Ho, Qb, GIC, GHP, Apr, Bpr, ghC, giP, Mg,
                                  cwhb, pwib, wr0b, wr1b,
                                  gCbh + l * DH3, gPbi + l * DH3,
                                  gatw, gatb, l, adj, smask);
  }

  // MLP head
  k_concat<<<(BN * INDIM + 255) / 256, 256, 0, stream>>>(H0, H1o, H2o, fused, Hcatb);
  gemm(Hcatb, INDIM, w0b, INDIM, b0, nullptr, h1, DH, BN, DH, DH, INDIM, 1);
  cvt(h1, h1bf, (size_t)BN * DH);
  gemm(h1bf, DH, w1b, DH, b1, nullptr, h2, DH, BN, DH, DH, DH, 1);
  cvt(h2, h2bf, (size_t)BN * DH);
  gemm1(h2bf, DH, w2b, DH, b2, lg, NCc, BN, 16, NCc, DH);
  k_logsoftmax<<<(BN + 255) / 256, 256, 0, stream>>>(lg, (float*)d_out, BN);
}
